// ThermalAttention_25958782337787
// MI455X (gfx1250) — matrix operand order repaired and checked by reading, not compiled, not run on hardware
//
#include <hip/hip_runtime.h>

// ThermalAttention for MI455X (gfx1250).
// bf16 WMMA everywhere; async-DMA (GLOBAL_LOAD_ASYNC_TO_LDS) tile staging.
// B=4, S=2048, C=1024, H=16, D=64.

#define DEVI __device__ __forceinline__

typedef __attribute__((ext_vector_type(16))) __bf16 v16bf;
typedef __attribute__((ext_vector_type(8)))  float  v8f;

union BF16Frag { unsigned short u[16]; v16bf v; };

DEVI unsigned short f2bf(float f) {
  unsigned u = __builtin_bit_cast(unsigned, f);
  unsigned r = u + 0x7FFFu + ((u >> 16) & 1u);   // round-to-nearest-even
  return (unsigned short)(r >> 16);
}

DEVI v8f zero8() { v8f z; for (int i = 0; i < 8; ++i) z[i] = 0.0f; return z; }

DEVI v8f wmma_bf16(const BF16Frag& a, const BF16Frag& b, v8f c) {
  // D = A(16x32 bf16) * B(32x16 bf16) + C(16x16 f32)
  return __builtin_amdgcn_wmma_f32_16x16x32_bf16(
      false, a.v, false, b.v, (short)0, c, false, false);
}

// ---- async global->LDS copy (CDNA5 GLOBAL_LOAD_ASYNC_TO_LDS_B128) ----------
#if defined(__has_builtin)
#if __has_builtin(__builtin_amdgcn_global_load_async_to_lds_b128) && \
    __has_builtin(__builtin_amdgcn_s_wait_asynccnt)
#define ASYNC_LDS 1
#endif
#endif
#ifndef ASYNC_LDS
#define ASYNC_LDS 0
#endif

typedef int i32x4 __attribute__((vector_size(16)));
typedef __attribute__((address_space(1))) i32x4 as1_i32x4;
typedef __attribute__((address_space(3))) i32x4 as3_i32x4;

DEVI void cp_lds_b128(void* lds, const void* gmem) {
#if ASYNC_LDS
  __builtin_amdgcn_global_load_async_to_lds_b128(
      (as1_i32x4*)(uintptr_t)gmem, (as3_i32x4*)(uintptr_t)lds, 0, 0);
#else
  *(uint4*)lds = *(const uint4*)gmem;
#endif
}

DEVI void wait_async() {
#if ASYNC_LDS
  __builtin_amdgcn_s_wait_asynccnt(0);
#endif
}

// ---------------------------------------------------------------------------
// Precision-convert kernels (run once per call, outside the hot GEMM loops).
// ---------------------------------------------------------------------------
__global__ __launch_bounds__(256) void cvt_kernel(
    const float* __restrict__ src, unsigned short* __restrict__ dst, int n) {
  const int i = blockIdx.x * 256 + threadIdx.x;
  if (i < n) dst[i] = f2bf(src[i]);
}

// dst[n*K + k] = bf16(src[k*N + n])  (emit transposed bf16 weights)
__global__ __launch_bounds__(256) void cvt_transpose_kernel(
    const float* __restrict__ src, unsigned short* __restrict__ dst,
    int K, int N) {
  const int i = blockIdx.x * 256 + threadIdx.x;
  if (i < N * K) {
    const int n = i / K, k = i - n * K;
    dst[i] = f2bf(src[(size_t)k * N + n]);
  }
}

__global__ void init_ent_kernel(float* __restrict__ entOut) {
  if (threadIdx.x < 4) entOut[threadIdx.x] = 0.0f;
}

// ---------------------------------------------------------------------------
// Kernel 1: qkv = x @ w_qkv (bf16 WMMA), per-head LayerNorm on q,k.
// q,k stored [B,H,S,D]; v stored TRANSPOSED [B,H,D,S] (feeds contiguous
// B-fragments in the attention kernel).
// Grid: (64, 48). Block: 256 (8 waves). Block tile 128(M) x 64(N), K-step 32.
// ---------------------------------------------------------------------------
__global__ __launch_bounds__(256) void qkv_ln_kernel(
    const unsigned short* __restrict__ xbf,   // [8192][1024] bf16
    const unsigned short* __restrict__ wT,    // [3072][1024] bf16 (w_qkv^T)
    const float* __restrict__ q_scale, const float* __restrict__ q_bias,
    const float* __restrict__ k_scale, const float* __restrict__ k_bias,
    unsigned short* __restrict__ qB, unsigned short* __restrict__ kB,
    unsigned short* __restrict__ vB)
{
  __shared__ __align__(16) unsigned short As[128][40];   // 128 x 32 (+pad)
  __shared__ __align__(16) unsigned short Bst[64][40];   // 64(n) x 32(k) (+pad)
  __shared__ __align__(16) float          Cs[128][68];   // 128 x 64 f32 (+pad)

  const int t    = threadIdx.x;
  const int lane = t & 31;
  const int wave = t >> 5;
  const int wm   = wave >> 1;      // 0..3 : M offset 32*wm
  const int wn   = wave & 1;       // 0..1 : N offset 32*wn
  const int l15  = lane & 15;
  const int half = lane >> 4;
  const int mBase = blockIdx.x * 128;
  const int nBase = blockIdx.y * 64;

  v8f c[2][2];
  for (int i = 0; i < 2; ++i)
    for (int j = 0; j < 2; ++j) c[i][j] = zero8();

  for (int kt = 0; kt < 1024; kt += 32) {
    // ---- async-stage A (128 rows x 64B) and B^T (64 rows x 64B) tiles ----
    for (int it = 0; it < 2; ++it) {
      const int cid = t + it * 256;           // 512 chunks of 16B
      const int row = cid >> 2;
      const int cc  = (cid & 3) * 8;
      cp_lds_b128(&As[row][cc], xbf + (size_t)(mBase + row) * 1024 + kt + cc);
    }
    {
      const int row = t >> 2;                 // 256 chunks of 16B
      const int cc  = (t & 3) * 8;
      cp_lds_b128(&Bst[row][cc], wT + (size_t)(nBase + row) * 1024 + kt + cc);
      if (kt + 32 < 1024)
        __builtin_prefetch(wT + (size_t)(nBase + row) * 1024 + kt + 32 + cc, 0, 1);
    }
    wait_async();
    __syncthreads();

    // ---- fragments (all 16B-contiguous LDS reads) + WMMA ----
    BF16Frag a[2], b[2];
    for (int tm = 0; tm < 2; ++tm) {
      const int row = wm * 32 + tm * 16 + l15;
      for (int i = 0; i < 8; ++i) {
        a[tm].u[i]     = As[row][half * 8 + i];        // K 0..7 / 8..15
        a[tm].u[8 + i] = As[row][16 + half * 8 + i];   // K 16..23 / 24..31
      }
    }
    for (int tn = 0; tn < 2; ++tn) {
      const int col = wn * 32 + tn * 16 + l15;
      for (int i = 0; i < 8; ++i) {
        b[tn].u[i]     = Bst[col][half * 8 + i];        // K 0..7 / 8..15
        b[tn].u[8 + i] = Bst[col][16 + half * 8 + i];   // K 16..23 / 24..31
      }
    }
    for (int tm = 0; tm < 2; ++tm)
      for (int tn = 0; tn < 2; ++tn)
        c[tm][tn] = wmma_bf16(a[tm], b[tn], c[tm][tn]);
    __syncthreads();
  }

  // ---- C -> LDS ----
  for (int tm = 0; tm < 2; ++tm)
    for (int tn = 0; tn < 2; ++tn)
      for (int j = 0; j < 8; ++j)
        Cs[wm * 32 + tm * 16 + half * 8 + j][wn * 32 + tn * 16 + l15] =
            c[tm][tn][j];
  __syncthreads();

  // ---- LayerNorm epilogue, one row per thread ----
  if (t < 128) {
    const int row  = t;
    const int gRow = mBase + row;
    const int bI   = gRow >> 11;
    const int sI   = gRow & 2047;
    const int which = blockIdx.y >> 4;  // 0=q 1=k 2=v
    const int h     = blockIdx.y & 15;
    if (which == 2) {
      // v: transposed layout [B,H,D,S]
      const size_t o = (size_t)(bI * 16 + h) * 64 * 2048 + sI;
      for (int d = 0; d < 64; ++d)
        vB[o + (size_t)d * 2048] = f2bf(Cs[row][d]);
    } else {
      unsigned short* dst = (which == 0) ? qB : kB;
      const float* sc = (which == 0) ? q_scale : k_scale;
      const float* bi = (which == 0) ? q_bias  : k_bias;
      float s1 = 0.f, s2 = 0.f;
      for (int d = 0; d < 64; ++d) { const float v = Cs[row][d]; s1 += v; s2 += v * v; }
      const float mu  = s1 * (1.0f / 64.0f);
      const float var = s2 * (1.0f / 64.0f) - mu * mu;
      const float inv = rsqrtf(var + 1e-5f);
      const size_t o = ((size_t)(bI * 16 + h) * 2048 + sI) * 64;
      for (int d = 0; d < 64; ++d)
        dst[o + d] = f2bf((Cs[row][d] - mu) * inv * sc[d] + bi[d]);
    }
  }
}

// ---------------------------------------------------------------------------
// Kernel 2: flash attention + online entropy per (b,h,64-row q tile).
// Block: 128 threads (4 waves); wave w owns q rows [16w,16w+16) x all keys.
// Grid: (32, 16, 4).
// ---------------------------------------------------------------------------
__global__ __launch_bounds__(128) void flash_kernel(
    const unsigned short* __restrict__ qB,   // [B,H,S,D]
    const unsigned short* __restrict__ kB,   // [B,H,S,D]
    const unsigned short* __restrict__ vB,   // [B,H,D,S] (transposed)
    unsigned short* __restrict__ attO,       // [B,S,H*D] bf16
    float* __restrict__ entOut)
{
  __shared__ __align__(16) unsigned short Qs [64][72];
  __shared__ __align__(16) unsigned short Kt [64][72];   // [key][d]
  __shared__ __align__(16) unsigned short VtT[64][72];   // [d][key]
  __shared__ __align__(16) unsigned short Ps [64][72];   // [qrow][key]
  __shared__ float blkEnt;

  const int t    = threadIdx.x;
  const int lane = t & 31;
  const int w    = t >> 5;
  const int l15  = lane & 15;
  const int half = lane >> 4;
  const int qBase = blockIdx.x * 64;
  const int h     = blockIdx.y;
  const int b     = blockIdx.z;
  const size_t headOff  = (size_t)(b * 16 + h) * 2048 * 64;  // q/k base
  const size_t vHeadOff = (size_t)(b * 16 + h) * 64 * 2048;  // v^T base

  if (t == 0) blkEnt = 0.f;

  // ---- Q tile -> LDS (async) ----
  {
    const int row = t >> 1, hf = t & 1;
    for (int j = 0; j < 4; ++j)
      cp_lds_b128(&Qs[row][hf * 32 + j * 8],
                  qB + headOff + (size_t)(qBase + row) * 64 + hf * 32 + j * 8);
  }
  wait_async();
  __syncthreads();

  BF16Frag aq[2];
  for (int kc = 0; kc < 2; ++kc) {
    const int row = w * 16 + l15;
    for (int i = 0; i < 8; ++i) {
      aq[kc].u[i]     = Qs[row][kc * 32 + half * 8 + i];
      aq[kc].u[8 + i] = Qs[row][kc * 32 + 16 + half * 8 + i];
    }
  }

  float m_[8], Z_[8], T_[8];
  for (int j = 0; j < 8; ++j) { m_[j] = -1e30f; Z_[j] = 0.f; T_[j] = 0.f; }
  v8f o[4];
  for (int nf = 0; nf < 4; ++nf) o[nf] = zero8();

  for (int kt2 = 0; kt2 < 2048; kt2 += 64) {
    __syncthreads();   // previous iteration readers done
    {
      const int row = t >> 1, hf = t & 1;
      for (int j = 0; j < 4; ++j) {
        cp_lds_b128(&Kt[row][hf * 32 + j * 8],
                    kB + headOff + (size_t)(kt2 + row) * 64 + hf * 32 + j * 8);
        cp_lds_b128(&VtT[row][hf * 32 + j * 8],
                    vB + vHeadOff + (size_t)row * 2048 + kt2 + hf * 32 + j * 8);
      }
    }
    wait_async();
    __syncthreads();

    // ---- logits = (Q K^T) / sqrt(D) ----
    v8f lg[4];
    for (int nf = 0; nf < 4; ++nf) {
      lg[nf] = zero8();
      for (int kc = 0; kc < 2; ++kc) {
        BF16Frag bk;   // B[k=d][n=key] = K[key][d]
        for (int i = 0; i < 8; ++i) {
          bk.u[i]     = Kt[nf * 16 + l15][kc * 32 + half * 8 + i];
          bk.u[8 + i] = Kt[nf * 16 + l15][kc * 32 + 16 + half * 8 + i];
        }
        lg[nf] = wmma_bf16(aq[kc], bk, lg[nf]);
      }
      for (int j = 0; j < 8; ++j) lg[nf][j] *= 0.125f;
    }

    // ---- online softmax (row reductions within 16-lane halves) ----
    float scl[8];
    for (int j = 0; j < 8; ++j) {
      float v = fmaxf(fmaxf(lg[0][j], lg[1][j]), fmaxf(lg[2][j], lg[3][j]));
      v = fmaxf(v, __shfl_xor(v, 1, 32));
      v = fmaxf(v, __shfl_xor(v, 2, 32));
      v = fmaxf(v, __shfl_xor(v, 4, 32));
      v = fmaxf(v, __shfl_xor(v, 8, 32));
      const float mn = fmaxf(m_[j], v);
      scl[j] = __expf(m_[j] - mn);
      m_[j]  = mn;
      Z_[j] *= scl[j];
      T_[j] *= scl[j];
    }
    for (int nf = 0; nf < 4; ++nf)
      for (int j = 0; j < 8; ++j) o[nf][j] *= scl[j];

    float za[8], ta[8];
    for (int j = 0; j < 8; ++j) { za[j] = 0.f; ta[j] = 0.f; }
    for (int nf = 0; nf < 4; ++nf)
      for (int j = 0; j < 8; ++j) {
        const float p = __expf(lg[nf][j] - m_[j]);
        za[j] += p;
        ta[j] += p * lg[nf][j];
        Ps[w * 16 + half * 8 + j][nf * 16 + l15] = f2bf(p);
      }
    for (int j = 0; j < 8; ++j) {
      float z = za[j], tt = ta[j];
      z += __shfl_xor(z, 1, 32); tt += __shfl_xor(tt, 1, 32);
      z += __shfl_xor(z, 2, 32); tt += __shfl_xor(tt, 2, 32);
      z += __shfl_xor(z, 4, 32); tt += __shfl_xor(tt, 4, 32);
      z += __shfl_xor(z, 8, 32); tt += __shfl_xor(tt, 8, 32);
      Z_[j] += z; T_[j] += tt;
    }
    __syncthreads();

    // ---- O += P V  (A = P rows, B = V^T rows -> contiguous reads) ----
    for (int nf = 0; nf < 4; ++nf) {
      for (int kc = 0; kc < 2; ++kc) {
        BF16Frag ap, bv;
        const int row = w * 16 + l15;
        for (int i = 0; i < 8; ++i) {
          ap.u[i]     = Ps[row][kc * 32 + half * 8 + i];
          ap.u[8 + i] = Ps[row][kc * 32 + 16 + half * 8 + i];
        }
        for (int i = 0; i < 8; ++i) {
          bv.u[i]     = VtT[nf * 16 + l15][kc * 32 + half * 8 + i];
          bv.u[8 + i] = VtT[nf * 16 + l15][kc * 32 + 16 + half * 8 + i];
        }
        o[nf] = wmma_bf16(ap, bv, o[nf]);
      }
    }
  }

  // ---- epilogue: normalize O -> bf16 [B,S,H*D]; entropy accumulation ----
  float entSum = 0.f;
  for (int j = 0; j < 8; ++j) {
    const float zin = 1.0f / Z_[j];
    const int row = qBase + w * 16 + half * 8 + j;
    for (int nf = 0; nf < 4; ++nf) {
      const int col = h * 64 + nf * 16 + l15;
      attO[((size_t)(b * 2048 + row)) * 1024 + col] = f2bf(o[nf][j] * zin);
    }
    if (l15 == 0) entSum += m_[j] + logf(Z_[j]) - T_[j] * zin;
  }
  if (l15 == 0) atomicAdd(&blkEnt, entSum);
  __syncthreads();
  if (t == 0) {
    const float norm = 1.0f / (11.090354888959125f * 16.0f * 2048.0f); // 1/(ln(65536)*H*S)
    atomicAdd(entOut + b, blkEnt * norm);
  }
}

// ---------------------------------------------------------------------------
// Kernel 3: out = attO @ w_proj  (bf16 x bf16^T -> f32)
// Grid: (64, 16). Block: 256. Direct f32 stores from fragments.
// ---------------------------------------------------------------------------
__global__ __launch_bounds__(256) void proj_kernel(
    const unsigned short* __restrict__ attO,   // [8192][1024] bf16
    const unsigned short* __restrict__ wT,     // [1024][1024] bf16 (w_proj^T)
    float* __restrict__ out)
{
  __shared__ __align__(16) unsigned short As[128][40];
  __shared__ __align__(16) unsigned short Bst[64][40];

  const int t    = threadIdx.x;
  const int lane = t & 31;
  const int wave = t >> 5;
  const int wm   = wave >> 1;
  const int wn   = wave & 1;
  const int l15  = lane & 15;
  const int half = lane >> 4;
  const int mBase = blockIdx.x * 128;
  const int nBase = blockIdx.y * 64;

  v8f c[2][2];
  for (int i = 0; i < 2; ++i)
    for (int j = 0; j < 2; ++j) c[i][j] = zero8();

  for (int kt = 0; kt < 1024; kt += 32) {
    for (int it = 0; it < 2; ++it) {
      const int cid = t + it * 256;
      const int row = cid >> 2;
      const int cc  = (cid & 3) * 8;
      cp_lds_b128(&As[row][cc], attO + (size_t)(mBase + row) * 1024 + kt + cc);
    }
    {
      const int row = t >> 2;
      const int cc  = (t & 3) * 8;
      cp_lds_b128(&Bst[row][cc], wT + (size_t)(nBase + row) * 1024 + kt + cc);
    }
    wait_async();
    __syncthreads();

    BF16Frag a[2], b[2];
    for (int tm = 0; tm < 2; ++tm) {
      const int row = wm * 32 + tm * 16 + l15;
      for (int i = 0; i < 8; ++i) {
        a[tm].u[i]     = As[row][half * 8 + i];
        a[tm].u[8 + i] = As[row][16 + half * 8 + i];
      }
    }
    for (int tn = 0; tn < 2; ++tn) {
      const int col = wn * 32 + tn * 16 + l15;
      for (int i = 0; i < 8; ++i) {
        b[tn].u[i]     = Bst[col][half * 8 + i];
        b[tn].u[8 + i] = Bst[col][16 + half * 8 + i];
      }
    }
    for (int tm = 0; tm < 2; ++tm)
      for (int tn = 0; tn < 2; ++tn)
        c[tm][tn] = wmma_bf16(a[tm], b[tn], c[tm][tn]);
    __syncthreads();
  }

  for (int tm = 0; tm < 2; ++tm)
    for (int tn = 0; tn < 2; ++tn)
      for (int j = 0; j < 8; ++j) {
        const int row = mBase + wm * 32 + tm * 16 + half * 8 + j;
        const int col = nBase + wn * 32 + tn * 16 + l15;
        out[(size_t)row * 1024 + col] = c[tm][tn][j];
      }
}

// ---------------------------------------------------------------------------
extern "C" void kernel_launch(void* const* d_in, const int* in_sizes, int n_in,
                              void* d_out, int out_size, void* d_ws, size_t ws_size,
                              hipStream_t stream) {
  (void)in_sizes; (void)n_in; (void)out_size; (void)ws_size;
  const float* x       = (const float*)d_in[0];
  const float* w_qkv   = (const float*)d_in[1];
  const float* w_proj  = (const float*)d_in[2];
  const float* q_scale = (const float*)d_in[3];
  const float* q_bias  = (const float*)d_in[4];
  const float* k_scale = (const float*)d_in[5];
  const float* k_bias  = (const float*)d_in[6];

  float* out    = (float*)d_out;
  float* entOut = out + (size_t)8192 * 1024;   // 4 entropy slots after [B,S,C]

  const size_t TEN = (size_t)4 * 16 * 2048 * 64;   // 8,388,608 elems
  unsigned short* qB     = (unsigned short*)d_ws;
  unsigned short* kB     = qB + TEN;
  unsigned short* vB     = kB + TEN;               // transposed [B,H,D,S]
  unsigned short* attO   = vB + TEN;
  unsigned short* xbf    = attO + TEN;             // 8,388,608
  unsigned short* wqkvT  = xbf + TEN;              // 3,145,728
  unsigned short* wprojT = wqkvT + (size_t)3072 * 1024;  // 1,048,576

  cvt_kernel<<<dim3(32768), dim3(256), 0, stream>>>(x, xbf, 8388608);
  cvt_transpose_kernel<<<dim3(12288), dim3(256), 0, stream>>>(
      w_qkv, wqkvT, 1024, 3072);
  cvt_transpose_kernel<<<dim3(4096), dim3(256), 0, stream>>>(
      w_proj, wprojT, 1024, 1024);
  init_ent_kernel<<<dim3(1), dim3(32), 0, stream>>>(entOut);

  qkv_ln_kernel<<<dim3(64, 48), dim3(256), 0, stream>>>(
      xbf, wqkvT, q_scale, q_bias, k_scale, k_bias, qB, kB, vB);
  flash_kernel<<<dim3(32, 16, 4), dim3(128), 0, stream>>>(
      qB, kB, vB, attO, entOut);
  proj_kernel<<<dim3(64, 16), dim3(256), 0, stream>>>(attO, wprojT, out);
}
